// GATNetHeadsChanged4LayersWithNonlinearity_31628139168041
// MI455X (gfx1250) — matrix operand order repaired and checked by reading, not compiled, not run on hardware
//
#include <hip/hip_runtime.h>
#include <hip/hip_bf16.h>
#include <math.h>

typedef __attribute__((ext_vector_type(16))) __bf16          v16bf;
typedef __attribute__((ext_vector_type(16))) unsigned short  v16us;
typedef __attribute__((ext_vector_type(8)))  unsigned short  v8us;
typedef __attribute__((ext_vector_type(8)))  float           v8f;

// ---------- helpers ----------
__device__ __forceinline__ unsigned short f2bf(float f) {
    unsigned u = __float_as_uint(f);
    u += 0x7FFFu + ((u >> 16) & 1u);          // round-to-nearest-even
    return (unsigned short)(u >> 16);
}
__device__ __forceinline__ unsigned mapOrd(float f) {      // order-preserving float->uint
    unsigned b = __float_as_uint(f);
    return (b & 0x80000000u) ? ~b : (b | 0x80000000u);
}
__device__ __forceinline__ float unmapOrd(unsigned u) {
    unsigned b = (u & 0x80000000u) ? (u ^ 0x80000000u) : ~u;
    return __uint_as_float(b);
}
__device__ __forceinline__ float wred(float s) {           // wave32 sum
    #pragma unroll
    for (int o = 16; o > 0; o >>= 1) s += __shfl_xor(s, o, 32);
    return s;
}
__device__ __forceinline__ float lrelu02(float x) { return x > 0.f ? x : 0.2f * x; }

struct AFrag { v8us lo; v8us hi; };           // 32B, bit_cast -> v16bf

// ---------- f32 -> bf16 elementwise ----------
__global__ void k_convert(const float* __restrict__ in, unsigned short* __restrict__ out, int total) {
    int t = blockIdx.x * blockDim.x + threadIdx.x;
    if (t < total) out[t] = f2bf(in[t]);
}

// ---------- weight prep: f32 [K,Nc] -> bf16 transposed [Nc,K] ----------
__global__ void k_prep_wt(const float* __restrict__ W, unsigned short* __restrict__ Wt,
                          int K, int Nc) {
    int t = blockIdx.x * blockDim.x + threadIdx.x;
    if (t >= K * Nc) return;
    int n = t / K, k = t % K;
    Wt[(size_t)n * K + k] = f2bf(W[(size_t)k * Nc + n]);
}

// ---------- WMMA bf16 GEMM: C[M,Nc] = A[M,K] @ W[K,Nc] (+bias) ----------
// A bf16 row-major, W pre-transposed bf16 [Nc][K]. One wave computes a 16x64 C slab
// (4 N-tiles) so the A fragment is reused across 4 WMMAs per K-step.
__global__ __launch_bounds__(256)
void k_gemm_bf16(const unsigned short* __restrict__ A, const unsigned short* __restrict__ Wt,
                 const float* __restrict__ bias, float* __restrict__ C,
                 int M, int K, int Nc) {
    const int lane  = threadIdx.x & 31;
    const int wave  = threadIdx.x >> 5;
    const int tile  = blockIdx.x * 8 + wave;
    const int tilesN4 = Nc >> 6;                       // slabs of 64 columns
    const int totTiles = (M >> 4) * tilesN4;
    if (tile >= totTiles) return;
    const int tm = tile / tilesN4, tn4 = tile % tilesN4;
    const int l = lane & 15, g = lane >> 4;
    const int arow_i = tm * 16 + l;

    v8f acc[4];
    #pragma unroll
    for (int j = 0; j < 4; ++j) {
        const float bv = bias ? bias[tn4 * 64 + j * 16 + l] : 0.0f;
        #pragma unroll
        for (int v = 0; v < 8; ++v) acc[j][v] = bv;
    }

    const unsigned short* arow = A + (size_t)arow_i * K;
    const unsigned short* brow[4];
    #pragma unroll
    for (int j = 0; j < 4; ++j)
        brow[j] = Wt + (size_t)(tn4 * 64 + j * 16 + l) * K;

    for (int k0 = 0; k0 < K; k0 += 32) {
        // A fragment: K(e) = (e/8)*16 + g*8 + (e%8) -> two contiguous 16B loads
        AFrag af;
        af.lo = *(const v8us*)(arow + k0 + g * 8);
        af.hi = *(const v8us*)(arow + k0 + 16 + g * 8);
        v16bf a = __builtin_bit_cast(v16bf, af);
        #pragma unroll
        for (int j = 0; j < 4; ++j) {
            // B fragment: K(e) = (e/8)*16 + g*8 + (e%8) -> two contiguous 16B loads
            AFrag bf_;
            bf_.lo = *(const v8us*)(brow[j] + k0 + g * 8);
            bf_.hi = *(const v8us*)(brow[j] + k0 + 16 + g * 8);
            v16bf b = __builtin_bit_cast(v16bf, bf_);
            acc[j] = __builtin_amdgcn_wmma_f32_16x16x32_bf16(
                         false, a, false, b, (short)0, acc[j], false, false);
        }
    }
    // C/D layout: row = tm*16 + g*8 + v, col = tn4*64 + j*16 + l
    #pragma unroll
    for (int j = 0; j < 4; ++j) {
        float* crow = C + (size_t)(tm * 16 + g * 8) * Nc + tn4 * 64 + j * 16 + l;
        #pragma unroll
        for (int v = 0; v < 8; ++v) crow[(size_t)v * Nc] = acc[j][v];
    }
}

// ---------- attention scores: a_src/a_dst [N,2] ----------
__global__ __launch_bounds__(256)
void k_att(const float* __restrict__ h0, const float* __restrict__ atts,
           const float* __restrict__ attd, float* __restrict__ as_,
           float* __restrict__ ad_, int N) {
    int w = (blockIdx.x * blockDim.x + threadIdx.x) >> 5;
    int lane = threadIdx.x & 31;
    if (w >= N) return;
    const float* hrow = h0 + (size_t)w * 512;
    float s0 = 0, s1 = 0, d0 = 0, d1 = 0;
    for (int c = lane; c < 256; c += 32) {
        float ha = hrow[c], hb = hrow[256 + c];
        s0 += ha * atts[c];       s1 += hb * atts[256 + c];
        d0 += ha * attd[c];       d1 += hb * attd[256 + c];
    }
    s0 = wred(s0); s1 = wred(s1); d0 = wred(d0); d1 = wred(d1);
    if (lane == 0) {
        as_[w * 2] = s0; as_[w * 2 + 1] = s1;
        ad_[w * 2] = d0; ad_[w * 2 + 1] = d1;
    }
}

__device__ __forceinline__ void edge_sd(const int* ei, int E, int e, int& s, int& d) {
    if (e < E) { s = ei[e]; d = ei[E + e]; } else { s = d = e - E; }
}

// ---------- edge softmax pass 1: per-dst max ----------
__global__ void k_edge_max(const int* __restrict__ ei, int E, int E2,
                           const float* __restrict__ as_, const float* __restrict__ ad_,
                           unsigned* __restrict__ m) {
    int t = blockIdx.x * blockDim.x + threadIdx.x;
    if (t >= E2 * 2) return;
    int h = t & 1, e = t >> 1, s, d;
    edge_sd(ei, E, e, s, d);
    float logit = lrelu02(as_[s * 2 + h] + ad_[d * 2 + h]);
    atomicMax(&m[d * 2 + h], mapOrd(logit));
}
// ---------- pass 2: exp + per-dst sum ----------
__global__ void k_edge_exp(const int* __restrict__ ei, int E, int E2,
                           const float* __restrict__ as_, const float* __restrict__ ad_,
                           const unsigned* __restrict__ m, float* __restrict__ z,
                           float* __restrict__ alpha) {
    int t = blockIdx.x * blockDim.x + threadIdx.x;
    if (t >= E2 * 2) return;
    int h = t & 1, e = t >> 1, s, d;
    edge_sd(ei, E, e, s, d);
    float logit = lrelu02(as_[s * 2 + h] + ad_[d * 2 + h]);
    float ex = __expf(logit - unmapOrd(m[d * 2 + h]));
    alpha[t] = ex;
    atomicAdd(&z[d * 2 + h], ex);
}
// ---------- pass 3: normalize ----------
__global__ void k_edge_div(const int* __restrict__ ei, int E, int E2,
                           const float* __restrict__ z, float* __restrict__ alpha) {
    int t = blockIdx.x * blockDim.x + threadIdx.x;
    if (t >= E2 * 2) return;
    int h = t & 1, e = t >> 1, s, d;
    edge_sd(ei, E, e, s, d);
    alpha[t] /= (z[d * 2 + h] + 1e-16f);
}
// ---------- pass 4: scatter messages (float4 gather + f32 atomics into L2) ----------
__global__ __launch_bounds__(256)
void k_scatter(const int* __restrict__ ei, int E, int E2,
               const float* __restrict__ alpha, const float* __restrict__ h0,
               float* __restrict__ agg) {
    int t = blockIdx.x * blockDim.x + threadIdx.x;
    if (t >= E2 * 128) return;                 // H*C/4 = 128 float4 chunks per edge
    int c4 = t & 63, h = (t >> 6) & 1, e = t >> 7;
    int s, d;
    edge_sd(ei, E, e, s, d);
    float a = alpha[(e << 1) | h];
    const float4 hv = *(const float4*)(h0 + (size_t)s * 512 + h * 256 + c4 * 4);
    float* o = agg + (size_t)d * 512 + h * 256 + c4 * 4;
    atomicAdd(o + 0, a * hv.x);
    atomicAdd(o + 1, a * hv.y);
    atomicAdd(o + 2, a * hv.z);
    atomicAdd(o + 3, a * hv.w);
}

// ---------- bias + relu -> bf16 (feeds next GEMM) ----------
__global__ void k_bias_relu_bf(const float* __restrict__ x, const float* __restrict__ b,
                               unsigned short* __restrict__ out, int total) {
    int t = blockIdx.x * blockDim.x + threadIdx.x;
    if (t >= total) return;
    out[t] = f2bf(fmaxf(x[t] + b[t & 511], 0.f));
}

// ---------- LayerNorm + activation; in-place f32 update + bf16 copy for next GEMM ----------
// mode 0: relu, mode 1: tanh(relu)
__global__ __launch_bounds__(256)
void k_ln_act(float* __restrict__ X, unsigned short* __restrict__ Xbf,
              const float* __restrict__ gam, const float* __restrict__ bet,
              int Nrows, int D, int mode) {
    int w = (blockIdx.x * blockDim.x + threadIdx.x) >> 5;
    int lane = threadIdx.x & 31;
    if (w >= Nrows) return;
    float* row = X + (size_t)w * D;
    unsigned short* rbf = Xbf + (size_t)w * D;
    float s = 0.f;
    for (int c = lane; c < D; c += 32) s += row[c];
    float mu = wred(s) / (float)D;
    float v = 0.f;
    for (int c = lane; c < D; c += 32) { float t = row[c] - mu; v += t * t; }
    float rs = rsqrtf(wred(v) / (float)D + 1e-5f);
    for (int c = lane; c < D; c += 32) {
        float y = (row[c] - mu) * rs * gam[c] + bet[c];
        y = fmaxf(y, 0.f);
        if (mode == 1) y = tanhf(y);
        row[c] = y;
        rbf[c] = f2bf(y);
    }
}

// ---------- final 64->3 linear, pack (x,y,z,|.|^2) ----------
__global__ void k_final(const float* __restrict__ h, const float* __restrict__ W3,
                        const float* __restrict__ b3, float4* __restrict__ hp, int N) {
    int n = blockIdx.x * blockDim.x + threadIdx.x;
    if (n >= N) return;
    const float* r = h + (size_t)n * 64;
    float o0 = b3[0], o1 = b3[1], o2 = b3[2];
    #pragma unroll 8
    for (int k = 0; k < 64; ++k) {
        float x = r[k];
        o0 += x * W3[k * 3]; o1 += x * W3[k * 3 + 1]; o2 += x * W3[k * 3 + 2];
    }
    hp[n] = make_float4(o0, o1, o2, o0 * o0 + o1 * o1 + o2 * o2);
}

// ---------- pairwise distances: 8x1024 tile per block, j staged in LDS ----------
__global__ __launch_bounds__(256)
void k_cdist(const float4* __restrict__ hp, float* __restrict__ out, int N) {
    __shared__ float4 tj[1024];
    const int tid = threadIdx.x;
    const int j0 = blockIdx.x * 1024;
    const int i0 = blockIdx.y * 8;
    #pragma unroll
    for (int r = 0; r < 4; ++r) {
        int j = j0 + tid + r * 256;
        tj[tid + r * 256] = (j < N) ? hp[j] : make_float4(0, 0, 0, 0);
    }
    __syncthreads();
    const int jj = tid * 4;
    if (j0 + jj >= N) return;                  // N % 4 == 0 -> whole float4 valid
    float4 b0 = tj[jj], b1 = tj[jj + 1], b2 = tj[jj + 2], b3 = tj[jj + 3];
    for (int i = 0; i < 8; ++i) {
        int ii = i0 + i;
        if (ii >= N) break;
        float4 a = hp[ii];
        float4 res;
        float d;
        d = a.w + b0.w - 2.f * (a.x * b0.x + a.y * b0.y + a.z * b0.z); res.x = sqrtf(fmaxf(d, 0.f));
        d = a.w + b1.w - 2.f * (a.x * b1.x + a.y * b1.y + a.z * b1.z); res.y = sqrtf(fmaxf(d, 0.f));
        d = a.w + b2.w - 2.f * (a.x * b2.x + a.y * b2.y + a.z * b2.z); res.z = sqrtf(fmaxf(d, 0.f));
        d = a.w + b3.w - 2.f * (a.x * b3.x + a.y * b3.y + a.z * b3.z); res.w = sqrtf(fmaxf(d, 0.f));
        *(float4*)(out + (size_t)ii * N + j0 + jj) = res;
    }
}

extern "C" void kernel_launch(void* const* d_in, const int* in_sizes, int n_in,
                              void* d_out, int out_size, void* d_ws, size_t ws_size,
                              hipStream_t stream) {
    const float* x       = (const float*)d_in[0];
    const int*   eidx    = (const int*)d_in[1];
    const float* W_gat   = (const float*)d_in[2];
    const float* att_src = (const float*)d_in[3];
    const float* att_dst = (const float*)d_in[4];
    const float* b_gat   = (const float*)d_in[5];
    const float* Wa      = (const float*)d_in[6];
    const float* ba      = (const float*)d_in[7];
    const float* ga      = (const float*)d_in[8];
    const float* bna     = (const float*)d_in[9];
    const float* W1      = (const float*)d_in[10];
    const float* b1      = (const float*)d_in[11];
    const float* g1      = (const float*)d_in[12];
    const float* bn1     = (const float*)d_in[13];
    const float* W2      = (const float*)d_in[14];
    const float* b2      = (const float*)d_in[15];
    const float* g2      = (const float*)d_in[16];
    const float* bn2     = (const float*)d_in[17];
    const float* W3      = (const float*)d_in[18];
    const float* b3      = (const float*)d_in[19];

    const int N  = in_sizes[0] / 512;
    const int E  = in_sizes[1] / 2;
    const int E2 = E + N;                      // with self loops

    char* wsb = (char*)d_ws;
    size_t off = 0;
    auto alloc = [&](size_t bytes) -> void* {
        void* p = wsb + off;
        off = (off + bytes + 255) & ~(size_t)255;
        return p;
    };
    unsigned short* wt_gat = (unsigned short*)alloc((size_t)512 * 512 * 2);
    unsigned short* wt_a   = (unsigned short*)alloc((size_t)256 * 512 * 2);
    unsigned short* wt_1   = (unsigned short*)alloc((size_t)128 * 256 * 2);
    unsigned short* wt_2   = (unsigned short*)alloc((size_t)64  * 128 * 2);
    unsigned short* x_bf   = (unsigned short*)alloc((size_t)N * 512 * 2);
    unsigned short* gat_bf = (unsigned short*)alloc((size_t)N * 512 * 2);
    unsigned short* h1_bf  = (unsigned short*)alloc((size_t)N * 256 * 2);
    unsigned short* h2_bf  = (unsigned short*)alloc((size_t)N * 128 * 2);
    float*    h0    = (float*)alloc((size_t)N * 512 * 4);
    float*    agg   = (float*)alloc((size_t)N * 512 * 4);
    float*    as_   = (float*)alloc((size_t)N * 2 * 4);
    float*    ad_   = (float*)alloc((size_t)N * 2 * 4);
    unsigned* mmax  = (unsigned*)alloc((size_t)N * 2 * 4);
    float*    zsum  = (float*)alloc((size_t)N * 2 * 4);
    float*    alpha = (float*)alloc((size_t)E2 * 2 * 4);
    float*    h1    = (float*)alloc((size_t)N * 256 * 4);
    float*    h2    = (float*)alloc((size_t)N * 128 * 4);
    float*    h2b   = (float*)alloc((size_t)N * 64 * 4);
    unsigned short* h2b_bf = (unsigned short*)alloc((size_t)N * 64 * 2); // unused sink (keeps k_ln_act uniform)
    float4*   hp    = (float4*)alloc((size_t)N * 16);

    hipMemsetAsync(agg, 0, (size_t)N * 512 * 4, stream);
    hipMemsetAsync(mmax, 0, (size_t)N * 2 * 4, stream);   // mapOrd-space minimum
    hipMemsetAsync(zsum, 0, (size_t)N * 2 * 4, stream);

    // weight prep (transpose -> bf16) and activation convert
    k_prep_wt<<<(512 * 512 + 255) / 256, 256, 0, stream>>>(W_gat, wt_gat, 512, 512);
    k_prep_wt<<<(512 * 256 + 255) / 256, 256, 0, stream>>>(Wa,    wt_a,   512, 256);
    k_prep_wt<<<(256 * 128 + 255) / 256, 256, 0, stream>>>(W1,    wt_1,   256, 128);
    k_prep_wt<<<(128 * 64  + 255) / 256, 256, 0, stream>>>(W2,    wt_2,   128, 64);
    k_convert<<<(N * 512 + 255) / 256, 256, 0, stream>>>(x, x_bf, N * 512);

    // h0 = x @ W_gat   (no bias: reference adds b_gat after aggregation)
    {
        int tiles = (N / 16) * (512 / 64);
        k_gemm_bf16<<<(tiles + 7) / 8, 256, 0, stream>>>(x_bf, wt_gat, nullptr, h0, N, 512, 512);
    }
    // attention logits, edge softmax, scatter
    k_att<<<(N + 7) / 8, 256, 0, stream>>>(h0, att_src, att_dst, as_, ad_, N);
    k_edge_max<<<(E2 * 2 + 255) / 256, 256, 0, stream>>>(eidx, E, E2, as_, ad_, mmax);
    k_edge_exp<<<(E2 * 2 + 255) / 256, 256, 0, stream>>>(eidx, E, E2, as_, ad_, mmax, zsum, alpha);
    k_edge_div<<<(E2 * 2 + 255) / 256, 256, 0, stream>>>(eidx, E, E2, zsum, alpha);
    {
        long long tot = (long long)E2 * 128;
        k_scatter<<<(int)((tot + 255) / 256), 256, 0, stream>>>(eidx, E, E2, alpha, h0, agg);
    }
    k_bias_relu_bf<<<(N * 512 + 255) / 256, 256, 0, stream>>>(agg, b_gat, gat_bf, N * 512);

    // MLP stack
    {
        int tiles = (N / 16) * (256 / 64);
        k_gemm_bf16<<<(tiles + 7) / 8, 256, 0, stream>>>(gat_bf, wt_a, ba, h1, N, 512, 256);
    }
    k_ln_act<<<(N + 7) / 8, 256, 0, stream>>>(h1, h1_bf, ga, bna, N, 256, 0);   // relu (leaky∘relu == relu)
    {
        int tiles = (N / 16) * (128 / 64);
        k_gemm_bf16<<<(tiles + 7) / 8, 256, 0, stream>>>(h1_bf, wt_1, b1, h2, N, 256, 128);
    }
    k_ln_act<<<(N + 7) / 8, 256, 0, stream>>>(h2, h2_bf, g1, bn1, N, 128, 1);   // tanh(relu)
    {
        int tiles = (N / 16) * (64 / 64);
        k_gemm_bf16<<<(tiles + 7) / 8, 256, 0, stream>>>(h2_bf, wt_2, b2, h2b, N, 128, 64);
    }
    k_ln_act<<<(N + 7) / 8, 256, 0, stream>>>(h2b, h2b_bf, g2, bn2, N, 64, 0);  // relu

    k_final<<<(N + 255) / 256, 256, 0, stream>>>(h2b, W3, b3, hp, N);

    dim3 cg((N + 1023) / 1024, (N + 7) / 8);
    k_cdist<<<cg, 256, 0, stream>>>(hp, (float*)d_out, N);
}
